// TCNAttentionBlock_33122787787259
// MI455X (gfx1250) — matrix operand order repaired and checked by reading, not compiled, not run on hardware
//
#include <hip/hip_runtime.h>

// ---------------- problem constants (reference shapes) ----------------
#define NB   8      // batch
#define SS   1024   // sequence
#define CC   512    // channels
#define HH   8      // heads
#define DD   64     // head dim
#define PP   16     // persistent tokens
#define SP   1040   // S + P
#define SPP  1056   // SP padded to multiple of 32 (WMMA K step)
#define K3   1536   // 3*C im2col K for conv

typedef __bf16 bf16_t;
typedef __attribute__((ext_vector_type(16))) __bf16 v16bf;
typedef __attribute__((ext_vector_type(8)))  float  v8f;

// ---------------- WMMA helpers (CDNA5 wave32, 16x16x32 bf16) ----------------
__device__ __forceinline__ v8f wmma_bf16f32(v16bf a, v16bf b, v8f c) {
  // (neg_a, A, neg_b, B, c_mod, C, reuse_a, reuse_b)
  return __builtin_amdgcn_wmma_f32_16x16x32_bf16(false, a, false, b, (short)0, c,
                                                 false, false);
}

// A fragment: 16x32 bf16, row-major A, lda in elements.
// lane L: M = L%16, half = L/16; elem j=0..15 -> K=16*half+j (contiguous run).
__device__ __forceinline__ v16bf load_a_frag(const bf16_t* __restrict__ A, int lda,
                                             int m0, int k0, int lane) {
  int m = m0 + (lane & 15);
  int half = lane >> 4;
  const bf16_t* p = A + (size_t)m * lda + k0 + 16 * half;
  v16bf r;
#pragma unroll
  for (int j = 0; j < 8; ++j) r[j] = p[j];
#pragma unroll
  for (int j = 0; j < 8; ++j) r[j + 8] = p[8 + j];
  return r;
}

// B fragment: 32x16 bf16 from B^T stored (N,K) row-major, ldb in elements.
// lane L: N = L%16, half = L/16; elem j -> K = 16*half + j (contiguous run).
__device__ __forceinline__ v16bf load_b_frag(const bf16_t* __restrict__ Bt, int ldb,
                                             int k0, int n0, int lane) {
  int n = n0 + (lane & 15);
  int half = lane >> 4;
  const bf16_t* p = Bt + (size_t)n * ldb + k0 + 16 * half;
  v16bf r;
#pragma unroll
  for (int j = 0; j < 16; ++j) r[j] = p[j];
  return r;
}

// ---------------- generic batched GEMM: D = A * B  (bf16 in, fp32 acc) --------
// blockDim = 32 (one wave). Each wave computes a (MT*16) x 16 output strip:
// one B fragment per k-step feeds MT WMMAs (register-level B reuse).
// grid = (Ntiles, Mtiles/MT, batch); batch z -> (b0,b1) = (z/H2, z%H2).
template <int MT>
__global__ void gemm_bf16_generic(const bf16_t* __restrict__ A,
                                  const bf16_t* __restrict__ Bt,
                                  bf16_t* __restrict__ Cb,
                                  float* __restrict__ Cf,
                                  int Kd, int lda, int ldb, int ldc, int H2,
                                  long sA0, long sA1, long sB0, long sB1,
                                  long sC0, long sC1) {
  int lane = threadIdx.x;
  int nt = blockIdx.x, mt0 = blockIdx.y * MT, z = blockIdx.z;
  long b0 = z / H2, b1 = z % H2;
  A += b0 * sA0 + b1 * sA1;
  Bt += b0 * sB0 + b1 * sB1;
  long coff = b0 * sC0 + b1 * sC1;

  v8f acc[MT];
#pragma unroll
  for (int i = 0; i < MT; ++i) acc[i] = (v8f){};

  for (int k0 = 0; k0 < Kd; k0 += 32) {
    v16bf b = load_b_frag(Bt, ldb, k0, nt * 16, lane);
    if (k0 + 32 < Kd)  // prefetch next B strip (global_prefetch_b8)
      __builtin_prefetch(Bt + (size_t)(nt * 16 + (lane & 15)) * ldb + k0 + 32, 0, 1);
#pragma unroll
    for (int i = 0; i < MT; ++i) {
      v16bf a = load_a_frag(A, lda, (mt0 + i) * 16, k0, lane);
      acc[i] = wmma_bf16f32(a, b, acc[i]);
    }
  }
  int col = nt * 16 + (lane & 15);
#pragma unroll
  for (int i = 0; i < MT; ++i) {
    int rbase = (mt0 + i) * 16 + 8 * (lane >> 4);
    if (Cf) {
#pragma unroll
      for (int r = 0; r < 8; ++r)
        Cf[coff + (size_t)(rbase + r) * ldc + col] = acc[i][r];
    }
    if (Cb) {
#pragma unroll
      for (int r = 0; r < 8; ++r)
        Cb[coff + (size_t)(rbase + r) * ldc + col] = (bf16_t)acc[i][r];
    }
  }
}

// fc GEMM with fused bias + residual(x) + dual store (fp32 x_res, bf16 into pad1 at t+2)
// 64x16 strip per wave. grid = (C/16, (N*S)/64)
__global__ void gemm_fc_kernel(const bf16_t* __restrict__ A,   // aob (N*S, C) bf16
                               const bf16_t* __restrict__ Bt,  // fc_w (O,C) bf16
                               const float* __restrict__ xin,  // x (N*S, C)
                               const float* __restrict__ bias, // fc_b
                               float* __restrict__ xres,       // (N*S, C)
                               bf16_t* __restrict__ pad1) {    // (N, S+2, C)
  int lane = threadIdx.x;
  int nt = blockIdx.x, mt0 = blockIdx.y * 4;
  v8f acc[4];
#pragma unroll
  for (int i = 0; i < 4; ++i) acc[i] = (v8f){};
  for (int k0 = 0; k0 < CC; k0 += 32) {
    v16bf b = load_b_frag(Bt, CC, k0, nt * 16, lane);
#pragma unroll
    for (int i = 0; i < 4; ++i) {
      v16bf a = load_a_frag(A, CC, (mt0 + i) * 16, k0, lane);
      acc[i] = wmma_bf16f32(a, b, acc[i]);
    }
  }
  int col = nt * 16 + (lane & 15);
  float bo = bias[col];
#pragma unroll
  for (int i = 0; i < 4; ++i) {
    int rbase = (mt0 + i) * 16 + 8 * (lane >> 4);
#pragma unroll
    for (int r = 0; r < 8; ++r) {
      int m = rbase + r;
      float v = xin[(size_t)m * CC + col] + acc[i][r] + bo;
      xres[(size_t)m * CC + col] = v;
      int n = m / SS, t = m % SS;
      pad1[((size_t)n * (SS + 2) + (t + 2)) * CC + col] = (bf16_t)v;
    }
  }
}

// causal conv as GEMM over contiguous im2col rows, fused bias+ReLU.
// grid = (C/16, S/64, N). outpad (bf16, padded) and/or outf (fp32) optional.
__global__ void gemm_conv_kernel(const bf16_t* __restrict__ Apad, // (N, S+2, C)
                                 const bf16_t* __restrict__ Bt,   // (O, 3C)
                                 const float* __restrict__ bias,
                                 bf16_t* __restrict__ outpad,     // (N, S+2, C) or null
                                 float* __restrict__ outf) {      // (N, S, C) or null
  int lane = threadIdx.x;
  int nt = blockIdx.x, mt0 = blockIdx.y * 4, n = blockIdx.z;
  const bf16_t* A = Apad + (size_t)n * (SS + 2) * CC;
  v8f acc[4];
#pragma unroll
  for (int i = 0; i < 4; ++i) acc[i] = (v8f){};
  for (int k0 = 0; k0 < K3; k0 += 32) {
    v16bf b = load_b_frag(Bt, K3, k0, nt * 16, lane);
    if (k0 + 32 < K3)
      __builtin_prefetch(Bt + (size_t)(nt * 16 + (lane & 15)) * K3 + k0 + 32, 0, 1);
#pragma unroll
    for (int i = 0; i < 4; ++i) {
      v16bf a = load_a_frag(A, CC, (mt0 + i) * 16, k0, lane);  // K spans rows t..t+2
      acc[i] = wmma_bf16f32(a, b, acc[i]);
    }
  }
  int col = nt * 16 + (lane & 15);
  float bo = bias[col];
#pragma unroll
  for (int i = 0; i < 4; ++i) {
    int rbase = (mt0 + i) * 16 + 8 * (lane >> 4);
#pragma unroll
    for (int r = 0; r < 8; ++r) {
      int t = rbase + r;
      float v = acc[i][r] + bo;
      v = v > 0.f ? v : 0.f;
      if (outpad) outpad[((size_t)n * (SS + 2) + (t + 2)) * CC + col] = (bf16_t)v;
      if (outf)  outf[((size_t)n * SS + t) * CC + col] = v;
    }
  }
}

// ---------------- elementwise / reduction kernels ----------------
__global__ void cvt_f32_to_bf16(const float* __restrict__ in, bf16_t* __restrict__ out,
                                long n) {
  long i = (long)blockIdx.x * blockDim.x + threadIdx.x;
  if (i < n) out[i] = (bf16_t)in[i];
}

// conv weights (O, I, 3) -> wt (O, kappa*C + i) bf16
__global__ void repack_convw(const float* __restrict__ w, bf16_t* __restrict__ wt) {
  long i = (long)blockIdx.x * blockDim.x + threadIdx.x;
  long total = (long)CC * CC * 3;
  if (i >= total) return;
  int o = i / (3 * CC); int rem = i % (3 * CC);
  int kk = rem / CC;    int ii = rem % CC;
  wt[i] = (bf16_t)w[((size_t)o * CC + ii) * 3 + kk];
}

// persistent keys into kb rows S..S+P-1  (kb layout (N, SP, H, D))
__global__ void fill_pk(const float* __restrict__ pk, bf16_t* __restrict__ kb) {
  long i = (long)blockIdx.x * blockDim.x + threadIdx.x;
  long total = (long)NB * PP * HH * DD;
  if (i >= total) return;
  int d = i % DD; long r = i / DD;
  int h = r % HH; r /= HH;
  int p = r % PP; int n = r / PP;
  kb[(((size_t)n * SP + SS + p) * HH + h) * DD + d] =
      (bf16_t)pk[((size_t)p * HH + h) * DD + d];
}

// vt[n][h][e][l] (bf16, l in [0,SPP)) from vb (N,S,H,D) bf16 + pv fp32, zero pad
__global__ void build_vt(const bf16_t* __restrict__ vb, const float* __restrict__ pv,
                         bf16_t* __restrict__ vt) {
  long i = (long)blockIdx.x * blockDim.x + threadIdx.x;
  long total = (long)NB * HH * DD * SPP;
  if (i >= total) return;
  int l = i % SPP; long r = i / SPP;
  int e = r % DD; r /= DD;
  int h = r % HH; int n = r / HH;
  bf16_t v;
  if (l < SS)      v = vb[(((size_t)n * SS + l) * HH + h) * DD + e];
  else if (l < SP) v = (bf16_t)pv[((size_t)(l - SS) * HH + h) * DD + e];
  else             v = (bf16_t)0.f;
  vt[i] = v;
}

// E2[n,h,q,k] = sum_g th_pre[h,g] * (E1[n,g,q,k] + alibi(g,q,k)), then mask -> -1e4
__global__ void thpre_bias_mask(const float* __restrict__ E1,
                                const int* __restrict__ mask,
                                const float* __restrict__ thp,
                                float* __restrict__ E2) {
  __shared__ float th[64];
  if (threadIdx.x < 64) th[threadIdx.x] = thp[threadIdx.x];
  __syncthreads();
  int kk = blockIdx.x * blockDim.x + threadIdx.x;
  int q = blockIdx.y, n = blockIdx.z;
  if (kk >= SP) return;
  size_t gs = (size_t)SS * SP;
  size_t base = (((size_t)n * HH) * SS + q) * (size_t)SP + kk;
  float dist = fabsf((float)q - (float)kk);
  int mval = (kk < SS) ? mask[((size_t)n * SS + q) * SS + kk] : 1;
  float in[8];
#pragma unroll
  for (int g = 0; g < 8; ++g) {
    float slope = exp2f(-(float)(g + 1));          // ALIBI_ALPHA=1, START_I=0
    float bias = (kk < SS) ? -dist * slope : 0.f;  // no bias over persistent keys
    in[g] = E1[base + (size_t)g * gs] + bias;
  }
#pragma unroll
  for (int h = 0; h < 8; ++h) {
    float a = 0.f;
#pragma unroll
    for (int g = 0; g < 8; ++g) a += th[h * 8 + g] * in[g];
    if (mval == 0) a = -1e4f;
    E2[base + (size_t)h * gs] = a;
  }
}

// in-place softmax over rows of length SP with scale 1/sqrt(C)
__global__ void softmax_rows(float* __restrict__ E, float scale) {
  __shared__ float red[256];
  size_t base = (size_t)blockIdx.x * SP;
  int tid = threadIdx.x;
  float m = -3.0e38f;
  for (int i = tid; i < SP; i += 256) m = fmaxf(m, E[base + i] * scale);
  red[tid] = m; __syncthreads();
  for (int s = 128; s > 0; s >>= 1) {
    if (tid < s) red[tid] = fmaxf(red[tid], red[tid + s]);
    __syncthreads();
  }
  float rmax = red[0]; __syncthreads();
  float sum = 0.f;
  for (int i = tid; i < SP; i += 256) {
    float e = expf(E[base + i] * scale - rmax);
    E[base + i] = e; sum += e;
  }
  red[tid] = sum; __syncthreads();
  for (int s = 128; s > 0; s >>= 1) {
    if (tid < s) red[tid] += red[tid + s];
    __syncthreads();
  }
  float inv = 1.f / red[0];
  for (int i = tid; i < SP; i += 256) E[base + i] *= inv;
}

// attb[n,h,q,kp] (bf16, SPP wide, zero-padded) = sum_g th_post[h,g]*attn[n,g,q,k]
__global__ void thpost_cvt(const float* __restrict__ E2,
                           const float* __restrict__ thp,
                           bf16_t* __restrict__ attb) {
  __shared__ float th[64];
  if (threadIdx.x < 64) th[threadIdx.x] = thp[threadIdx.x];
  __syncthreads();
  int kk = blockIdx.x * blockDim.x + threadIdx.x;
  int q = blockIdx.y, n = blockIdx.z;
  if (kk >= SPP) return;
  size_t gsO = (size_t)SS * SPP;
  size_t baseO = (((size_t)n * HH) * SS + q) * (size_t)SPP + kk;
  if (kk >= SP) {
#pragma unroll
    for (int h = 0; h < 8; ++h) attb[baseO + (size_t)h * gsO] = (bf16_t)0.f;
    return;
  }
  size_t gsI = (size_t)SS * SP;
  size_t baseI = (((size_t)n * HH) * SS + q) * (size_t)SP + kk;
  float in[8];
#pragma unroll
  for (int g = 0; g < 8; ++g) in[g] = E2[baseI + (size_t)g * gsI];
#pragma unroll
  for (int h = 0; h < 8; ++h) {
    float a = 0.f;
#pragma unroll
    for (int g = 0; g < 8; ++g) a += th[h * 8 + g] * in[g];
    attb[baseO + (size_t)h * gsO] = (bf16_t)a;
  }
}

// zero the 2 leading pad rows per batch of a (N, S+2, C) bf16 buffer
__global__ void zero_padrows(bf16_t* __restrict__ pad) {
  long i = (long)blockIdx.x * blockDim.x + threadIdx.x;
  long total = (long)NB * 2 * CC;
  if (i >= total) return;
  int n = i / (2 * CC); long rem = i % (2 * CC);
  pad[(size_t)n * (SS + 2) * CC + rem] = (bf16_t)0.f;
}

// out = LN(mask ? relu(h2 + x_res) : 0) over C; block=128, grid=N*S
__global__ void final_ln(const float* __restrict__ h2, const float* __restrict__ xres,
                         const int* __restrict__ mask, const float* __restrict__ g,
                         const float* __restrict__ b, float* __restrict__ out) {
  __shared__ float r1[128];
  __shared__ float r2[128];
  int row = blockIdx.x;
  int n = row / SS, t = row % SS;
  int cm = mask[((size_t)n * SS + t) * SS];  // mask[n,0,t,0]
  int tid = threadIdx.x;
  float vals[4];
  float sum = 0.f, sq = 0.f;
#pragma unroll
  for (int j = 0; j < 4; ++j) {
    int c = tid + j * 128;
    float v = h2[(size_t)row * CC + c] + xres[(size_t)row * CC + c];
    v = v > 0.f ? v : 0.f;
    if (cm == 0) v = 0.f;
    vals[j] = v; sum += v; sq += v * v;
  }
  r1[tid] = sum; r2[tid] = sq; __syncthreads();
  for (int s = 64; s > 0; s >>= 1) {
    if (tid < s) { r1[tid] += r1[tid + s]; r2[tid] += r2[tid + s]; }
    __syncthreads();
  }
  float mu = r1[0] * (1.f / CC);
  float var = r2[0] * (1.f / CC) - mu * mu;
  float inv = rsqrtf(var + 1e-5f);
#pragma unroll
  for (int j = 0; j < 4; ++j) {
    int c = tid + j * 128;
    out[(size_t)row * CC + c] = (vals[j] - mu) * inv * g[c] + b[c];
  }
}

// ---------------- host orchestration ----------------
extern "C" void kernel_launch(void* const* d_in, const int* in_sizes, int n_in,
                              void* d_out, int out_size, void* d_ws, size_t ws_size,
                              hipStream_t stream) {
  const float* x       = (const float*)d_in[0];
  const int*   mask    = (const int*)  d_in[1];
  const float* Wq      = (const float*)d_in[2];
  const float* Wk      = (const float*)d_in[3];
  const float* Wv      = (const float*)d_in[4];
  const float* pk      = (const float*)d_in[5];
  const float* pv      = (const float*)d_in[6];
  const float* th_pre  = (const float*)d_in[7];
  const float* th_post = (const float*)d_in[8];
  const float* fc_w    = (const float*)d_in[9];
  const float* fc_b    = (const float*)d_in[10];
  const float* c1w     = (const float*)d_in[11];
  const float* c1b     = (const float*)d_in[12];
  const float* c2w     = (const float*)d_in[13];
  const float* c2b     = (const float*)d_in[14];
  const float* ln_g    = (const float*)d_in[15];
  const float* ln_b    = (const float*)d_in[16];
  float* out = (float*)d_out;

  // workspace carve-up (~650 MB required)
  char* w = (char*)d_ws; size_t off = 0;
  auto alloc = [&](size_t bytes) -> void* {
    void* p = w + off; off += (bytes + 255) & ~(size_t)255; return p;
  };
  const long nsc = (long)NB * SS * CC;                 // 4,194,304
  bf16_t* xb   = (bf16_t*)alloc(nsc * 2);
  bf16_t* Wqb  = (bf16_t*)alloc(DD * DD * 2);
  bf16_t* Wkb  = (bf16_t*)alloc(DD * DD * 2);
  bf16_t* Wvb  = (bf16_t*)alloc(DD * DD * 2);
  bf16_t* fcwb = (bf16_t*)alloc((size_t)CC * CC * 2);
  bf16_t* w1t  = (bf16_t*)alloc((size_t)CC * K3 * 2);
  bf16_t* w2t  = (bf16_t*)alloc((size_t)CC * K3 * 2);
  bf16_t* qb   = (bf16_t*)alloc(nsc * 2);
  bf16_t* kb   = (bf16_t*)alloc((size_t)NB * SP * CC * 2);
  bf16_t* vb   = (bf16_t*)alloc(nsc * 2);
  bf16_t* vt   = (bf16_t*)alloc((size_t)NB * HH * DD * SPP * 2);
  float*  E1   = (float*) alloc((size_t)NB * HH * SS * SP * 4);   // 273 MB
  float*  E2   = (float*) alloc((size_t)NB * HH * SS * SP * 4);   // 273 MB
  bf16_t* attb = (bf16_t*)E1;  // overlay: E1 dead after thpre pass
  bf16_t* aob  = (bf16_t*)alloc(nsc * 2);
  float*  xres = (float*) alloc(nsc * 4);
  bf16_t* pad1 = (bf16_t*)alloc((size_t)NB * (SS + 2) * CC * 2);
  bf16_t* pad2 = (bf16_t*)alloc((size_t)NB * (SS + 2) * CC * 2);
  float*  h2   = (float*) alloc(nsc * 4);

  // --- precision conversions / weight repacks ---
  cvt_f32_to_bf16<<<(nsc + 255) / 256, 256, 0, stream>>>(x, xb, nsc);
  cvt_f32_to_bf16<<<16, 256, 0, stream>>>(Wq, Wqb, DD * DD);
  cvt_f32_to_bf16<<<16, 256, 0, stream>>>(Wk, Wkb, DD * DD);
  cvt_f32_to_bf16<<<16, 256, 0, stream>>>(Wv, Wvb, DD * DD);
  cvt_f32_to_bf16<<<(CC * CC + 255) / 256, 256, 0, stream>>>(fc_w, fcwb, CC * CC);
  repack_convw<<<(CC * CC * 3 + 255) / 256, 256, 0, stream>>>(c1w, w1t);
  repack_convw<<<(CC * CC * 3 + 255) / 256, 256, 0, stream>>>(c2w, w2t);

  // --- QKV projections (q = xh @ Wq^T per head; weight shared across heads) ---
  // q, v: flat (N*S*H, D) GEMM. k: batched per n into (N, SP, H, D) with row gap.
  {
    dim3 g(DD / 16, (NB * SS * HH) / 64, 1);
    gemm_bf16_generic<4><<<g, 32, 0, stream>>>(xb, Wqb, qb, nullptr,
        DD, DD, DD, DD, 1, 0, 0, 0, 0, 0, 0);
    gemm_bf16_generic<4><<<g, 32, 0, stream>>>(xb, Wvb, vb, nullptr,
        DD, DD, DD, DD, 1, 0, 0, 0, 0, 0, 0);
    dim3 gk(DD / 16, (SS * HH) / 64, NB);
    gemm_bf16_generic<4><<<gk, 32, 0, stream>>>(xb, Wkb, kb, nullptr,
        DD, DD, DD, DD, 1,
        (long)SS * CC, 0, 0, 0, (long)SP * CC, 0);
  }
  fill_pk<<<(NB * PP * HH * DD + 255) / 256, 256, 0, stream>>>(pk, kb);
  build_vt<<<((long)NB * HH * DD * SPP + 255) / 256, 256, 0, stream>>>(vb, pv, vt);

  // --- energy = q @ k^T per (n,h): (S,64) x (64,SP) -> fp32 E1 ---
  {
    dim3 g(SP / 16, SS / 64, NB * HH);
    gemm_bf16_generic<4><<<g, 32, 0, stream>>>(qb, kb, nullptr, E1,
        DD, CC, CC, SP, HH,
        (long)SS * CC, DD,                 // A: n-stride, h-stride into qb
        (long)SP * CC, DD,                 // B: n-stride, h-stride into kb
        (long)HH * SS * SP, (long)SS * SP);// C: n-stride, h-stride into E1
  }

  // --- ALiBi + talking-heads(pre) + mask -> E2 ; softmax ; talking-heads(post) ---
  {
    dim3 g((SP + 255) / 256, SS, NB);
    thpre_bias_mask<<<g, 256, 0, stream>>>(E1, mask, th_pre, E2);
  }
  softmax_rows<<<NB * HH * SS, 256, 0, stream>>>(E2, rsqrtf((float)CC));
  {
    dim3 g((SPP + 255) / 256, SS, NB);
    thpost_cvt<<<g, 256, 0, stream>>>(E2, th_post, attb);
  }

  // --- att_out = attn @ v per (n,h): (S,SPP) x (SPP,64) -> bf16 aob (N,S,H,D) ---
  {
    dim3 g(DD / 16, SS / 64, NB * HH);
    gemm_bf16_generic<4><<<g, 32, 0, stream>>>(attb, vt, aob, nullptr,
        SPP, SPP, SPP, CC, HH,
        (long)HH * SS * SPP, (long)SS * SPP,
        (long)HH * DD * SPP, (long)DD * SPP,
        (long)SS * CC, DD);
  }

  // --- fc + bias + residual -> xres (fp32) and pad1 (bf16, causal pad 2) ---
  zero_padrows<<<(NB * 2 * CC + 255) / 256, 256, 0, stream>>>(pad1);
  zero_padrows<<<(NB * 2 * CC + 255) / 256, 256, 0, stream>>>(pad2);
  {
    dim3 g(CC / 16, (NB * SS) / 64, 1);
    gemm_fc_kernel<<<g, 32, 0, stream>>>(aob, fcwb, x, fc_b, xres, pad1);
  }

  // --- two causal dilated convs as K=1536 GEMMs, fused bias+ReLU ---
  {
    dim3 g(CC / 16, SS / 64, NB);
    gemm_conv_kernel<<<g, 32, 0, stream>>>(pad1, w1t, c1b, pad2, nullptr);
    gemm_conv_kernel<<<g, 32, 0, stream>>>(pad2, w2t, c2b, nullptr, h2);
  }

  // --- relu(h2 + xres), mask, LayerNorm -> out ---
  final_ln<<<NB * SS, 128, 0, stream>>>(h2, xres, mask, ln_g, ln_b, out);

  (void)in_sizes; (void)n_in; (void)out_size; (void)ws_size;
}
